// GeneralConv_22479858827470
// MI455X (gfx1250) — matrix operand order repaired and checked by reading, not compiled, not run on hardware
//
#include <hip/hip_runtime.h>
#include <hip/hip_bf16.h>

#define N_NODES 100000
#define N_EDGES 600000
#define DD      128

typedef __attribute__((ext_vector_type(16))) __bf16 v16bf;
typedef __attribute__((ext_vector_type(8)))  __bf16 v8bf;
typedef __attribute__((ext_vector_type(4)))  __bf16 v4bf;
typedef __attribute__((ext_vector_type(8)))  float  v8f;
typedef __attribute__((ext_vector_type(4)))  float  v4f;

// padded LDS row strides (stride % 64 banks == 4 banks -> conflict-free)
#define WT_STR  264   // bf16 elems for 256-wide Wt^T rows
#define W1_STR  136   // bf16 elems for 128-wide weight rows
#define SB_STR  136   // bf16 elems, edge_t staging rows
#define SF_STR  132   // f32  elems, node staging rows

// native hardware converts (RNE), avoids scalar bit-twiddling VALU bloat
__device__ __forceinline__ __bf16 f2bf(float f){ return (__bf16)f; }
__device__ __forceinline__ float bf2f(__bf16 b){ return (float)b; }

__device__ __forceinline__ float gelu_exact(float v){
  return 0.5f * v * (1.0f + erff(v * 0.70710678118654752440f));
}
__device__ __forceinline__ unsigned fenc(float f){
  unsigned u = __float_as_uint(f);
  return (u & 0x80000000u) ? ~u : (u | 0x80000000u);
}
__device__ __forceinline__ float fdec(unsigned e){
  unsigned u = (e & 0x80000000u) ? (e & 0x7FFFFFFFu) : ~e;
  return __uint_as_float(u);
}

// ---- WMMA fragment builders (wave32, 16x16x32 bf16) ----
__device__ __forceinline__ v16bf fragA_bf(const __bf16* row, int k0, int half){
  v8bf lo = *(const v8bf*)(row + k0 + half*8);
  v8bf hi = *(const v8bf*)(row + k0 + 16 + half*8);
  v16bf r;
#pragma unroll
  for (int i = 0; i < 8; ++i){ r[i] = lo[i]; r[i+8] = hi[i]; }
  return r;
}
__device__ __forceinline__ v16bf fragA_f32(const float* row, int k0, int half){
  v4f a0 = *(const v4f*)(row + k0 + half*8);
  v4f a1 = *(const v4f*)(row + k0 + half*8 + 4);
  v4f b0 = *(const v4f*)(row + k0 + 16 + half*8);
  v4f b1 = *(const v4f*)(row + k0 + 16 + half*8 + 4);
  v4bf c0 = __builtin_convertvector(a0, v4bf);
  v4bf c1 = __builtin_convertvector(a1, v4bf);
  v4bf c2 = __builtin_convertvector(b0, v4bf);
  v4bf c3 = __builtin_convertvector(b1, v4bf);
  v16bf r;
#pragma unroll
  for (int i=0;i<4;++i){
    r[i]    = c0[i]; r[i+4]  = c1[i];
    r[i+8]  = c2[i]; r[i+12] = c3[i];
  }
  return r;
}
// B (32x16): lane holds B^T row n=lane&15; elems 0-7: K=k0+8*half+i, elems 8-15: K=k0+16+8*half+i
__device__ __forceinline__ v16bf fragB_bf(const __bf16* btrow, int k0, int half){
  v8bf lo = *(const v8bf*)(btrow + k0 + half*8);
  v8bf hi = *(const v8bf*)(btrow + k0 + 16 + half*8);
  v16bf r;
#pragma unroll
  for (int i=0;i<8;++i){ r[i]=lo[i]; r[i+8]=hi[i]; }
  return r;
}
__device__ __forceinline__ v8f wmma_bf16(v16bf a, v16bf b, v8f c){
  return __builtin_amdgcn_wmma_f32_16x16x32_bf16(false, a, false, b, (short)0, c, false, false);
}
// copy one dense bf16 row (n elems) into padded LDS row
__device__ __forceinline__ void copy_row_to_lds(__bf16* dst, const __bf16* src, int n){
#pragma unroll
  for (int j = 0; j < 16; ++j){
    if (j*8 < n) *(v8bf*)(dst + j*8) = *(const v8bf*)(src + j*8);
  }
}

// ---------------- kernel 1: weight prep (bf16 transposes) ----------------
__global__ __launch_bounds__(256)
void prep_w_kernel(const float* __restrict__ Wt, const float* __restrict__ Wq,
                   const float* __restrict__ Wk, const float* __restrict__ Wv,
                   __bf16* __restrict__ wtT, __bf16* __restrict__ wqT,
                   __bf16* __restrict__ wkb, __bf16* __restrict__ wvT){
  int i = blockIdx.x * 256 + threadIdx.x;
  if (i < 128*256){ int n = i >> 8, k = i & 255; wtT[i] = f2bf(Wt[k*128 + n]); }
  int j = i - 128*256;
  if (j >= 0 && j < 128*128){
    int n = j >> 7, k = j & 127;
    wqT[j] = f2bf(Wq[k*128 + n]);
    wvT[j] = f2bf(Wv[k*128 + n]);
    wkb[j] = f2bf(Wk[j]);           // used as B^T directly (P = Q @ Wk^T)
  }
}

// ---------------- kernel 2: LayerNorm -> h (bf16) ----------------
__global__ __launch_bounds__(256)
void ln_kernel(const float* __restrict__ x, const float* __restrict__ gamma,
               const float* __restrict__ beta, __bf16* __restrict__ hb){
  int wave = threadIdx.x >> 5, lane = threadIdx.x & 31;
  long node = (long)blockIdx.x * 8 + wave;
  if (node >= N_NODES) return;
  const float* xr = x + node * DD;
  v4f v = *(const v4f*)(xr + lane*4);
  float s  = v[0]+v[1]+v[2]+v[3];
  float s2 = v[0]*v[0]+v[1]*v[1]+v[2]*v[2]+v[3]*v[3];
#pragma unroll
  for (int o = 16; o > 0; o >>= 1){ s += __shfl_xor(s, o, 32); s2 += __shfl_xor(s2, o, 32); }
  float mu  = s * (1.0f/128.0f);
  float var = s2 * (1.0f/128.0f) - mu*mu;
  float rs  = rsqrtf(var + 1e-5f);
  v4f g = *(const v4f*)(gamma + lane*4);
  v4f b = *(const v4f*)(beta  + lane*4);
  v4f hv;
#pragma unroll
  for (int i=0;i<4;++i) hv[i] = (v[i]-mu)*rs*g[i] + b[i];
  v4bf o = __builtin_convertvector(hv, v4bf);
  *(v4bf*)(hb + node*DD + lane*4) = o;
}

// ------- kernel 3: per-node Q = h@Wq+bq ; c = bk·Q ; P = Q@Wk^T -------
// persistent blocks, Wq^T and Wk cached in LDS, padded staging
__global__ __launch_bounds__(256)
void node_qp_kernel(const __bf16* __restrict__ hb, const __bf16* __restrict__ wqT,
                    const float* __restrict__ bq, const __bf16* __restrict__ wkb,
                    const float* __restrict__ bk, float* __restrict__ P,
                    float* __restrict__ c){
  extern __shared__ __align__(16) char smem[];
  __bf16* Wq_l = (__bf16*)smem;                           // 128 rows * W1_STR
  __bf16* Wk_l = Wq_l + 128*W1_STR;                       // 128 rows * W1_STR
  float*  Sbase = (float*)(smem + 2*128*W1_STR*2);

  int wave = threadIdx.x >> 5, lane = threadIdx.x & 31;
  int tid = threadIdx.x;
  {  // cooperative weight copy: tid<128 -> Wq row, else Wk row
    int r = tid & 127;
    if (tid < 128) copy_row_to_lds(Wq_l + r*W1_STR, wqT + r*DD, DD);
    else           copy_row_to_lds(Wk_l + r*W1_STR, wkb + r*DD, DD);
  }
  __syncthreads();

  int m = lane & 15, half = lane >> 4;
  float* L = Sbase + wave * 16*SF_STR;

  for (int tile = blockIdx.x*8 + wave; tile < N_NODES/16; tile += gridDim.x*8){
    long nb = (long)tile * 16;
    const __bf16* hrow = hb + (nb + m) * DD;

    v8f acc[8] = {};
#pragma unroll
    for (int kt = 0; kt < 4; ++kt){
      int k0 = kt * 32;
      v16bf a = fragA_bf(hrow, k0, half);
#pragma unroll
      for (int nt = 0; nt < 8; ++nt)
        acc[nt] = wmma_bf16(a, fragB_bf(Wq_l + (nt*16 + m)*W1_STR, k0, half), acc[nt]);
    }
#pragma unroll
    for (int nt = 0; nt < 8; ++nt){
      int col = nt*16 + m; float bqc = bq[col];
#pragma unroll
      for (int r = 0; r < 8; ++r) L[(r + 8*half)*SF_STR + col] = acc[nt][r] + bqc;
    }
    __builtin_amdgcn_wave_barrier();

    if (lane < 16){                               // c[n] = bk . Q[n]
      float s = 0.f;
      for (int k = 0; k < DD; ++k) s += L[lane*SF_STR + k] * bk[k];
      c[nb + lane] = s;
    }
    __builtin_amdgcn_wave_barrier();

    v8f pacc[8] = {};
#pragma unroll
    for (int kt = 0; kt < 4; ++kt){
      int k0 = kt * 32;
      v16bf a = fragA_f32(L + m*SF_STR, k0, half);  // Q tile from LDS
#pragma unroll
      for (int nt = 0; nt < 8; ++nt)
        pacc[nt] = wmma_bf16(a, fragB_bf(Wk_l + (nt*16 + m)*W1_STR, k0, half), pacc[nt]);
    }
    __builtin_amdgcn_wave_barrier();
#pragma unroll
    for (int nt = 0; nt < 8; ++nt){
      int col = nt*16 + m;
#pragma unroll
      for (int r = 0; r < 8; ++r) L[(r + 8*half)*SF_STR + col] = pacc[nt][r];
    }
    __builtin_amdgcn_wave_barrier();
    float* out = P + nb * DD;
#pragma unroll
    for (int i = 0; i < 16; ++i)
      *(v4f*)(out + i*DD + lane*4) = *(const v4f*)(L + i*SF_STR + lane*4);
    __builtin_amdgcn_wave_barrier();
  }
}

// ------- kernel 4: per-edge t = gelu([h_src, ev] @ Wt + bt) (bf16 out) -------
// persistent blocks, Wt^T cached in LDS, padded staging
__global__ __launch_bounds__(256)
void edge_t_kernel(const __bf16* __restrict__ hb, const float* __restrict__ ev,
                   const int* __restrict__ src, const __bf16* __restrict__ wtT,
                   const float* __restrict__ bt, __bf16* __restrict__ t){
  extern __shared__ __align__(16) char smem[];
  __bf16* Wt_l = (__bf16*)smem;                           // 128 rows * WT_STR
  __bf16* Sbase = Wt_l + 128*WT_STR;

  int wave = threadIdx.x >> 5, lane = threadIdx.x & 31;
  int tid = threadIdx.x;
  {  // cooperative copy: 2 threads per row (256 elems each row)
    int r = tid >> 1, hf = tid & 1;
    copy_row_to_lds(Wt_l + r*WT_STR + hf*128, wtT + r*256 + hf*128, 128);
  }
  __syncthreads();

  int m = lane & 15, half = lane >> 4;
  __bf16* L = Sbase + wave * 16*SB_STR;

  for (int tile = blockIdx.x*8 + wave; tile < N_EDGES/16; tile += gridDim.x*8){
    long eb = (long)tile * 16;
    const __bf16* hrow  = hb + (long)src[eb + m] * DD;
    const float*  evrow = ev + (eb + m) * DD;

    v8f acc[8] = {};
#pragma unroll
    for (int kt = 0; kt < 4; ++kt){               // K 0..127 from h[src]
      int k0 = kt * 32;
      v16bf a = fragA_bf(hrow, k0, half);
#pragma unroll
      for (int nt = 0; nt < 8; ++nt)
        acc[nt] = wmma_bf16(a, fragB_bf(Wt_l + (nt*16 + m)*WT_STR, k0, half), acc[nt]);
    }
#pragma unroll
    for (int kt = 0; kt < 4; ++kt){               // K 128..255 from edge_vector
      int k0 = kt * 32;
      v16bf a = fragA_f32(evrow, k0, half);
#pragma unroll
      for (int nt = 0; nt < 8; ++nt)
        acc[nt] = wmma_bf16(a, fragB_bf(Wt_l + (nt*16 + m)*WT_STR, 128 + k0, half), acc[nt]);
    }

#pragma unroll
    for (int nt = 0; nt < 8; ++nt){
      int col = nt*16 + m; float btc = bt[col];
#pragma unroll
      for (int r = 0; r < 8; ++r)
        L[(r + 8*half)*SB_STR + col] = f2bf(gelu_exact(acc[nt][r] + btc));
    }
    __builtin_amdgcn_wave_barrier();
    __bf16* drow = t + eb * DD;
#pragma unroll
    for (int i = 0; i < 8; ++i){                  // two rows per iteration
      int row = i*2 + half;
      int co  = m * 8;
      *(v8bf*)(drow + row*DD + co) = *(const v8bf*)(L + row*SB_STR + co);
    }
    __builtin_amdgcn_wave_barrier();
  }
}

// ------- kernel 5: attn_e = (t.P[dst] + c[dst]) * et / sqrt(D); segment max -------
__global__ __launch_bounds__(256)
void edge_attn_kernel(const __bf16* __restrict__ t, const float* __restrict__ P,
                      const float* __restrict__ c, const int* __restrict__ dst,
                      const float* __restrict__ etype, float* __restrict__ attn,
                      unsigned* __restrict__ nmax){
  int wave = threadIdx.x >> 5, lane = threadIdx.x & 31;
  long e = (long)blockIdx.x * 8 + wave;
  if (e >= N_EDGES) return;
  int d = dst[e];
  v4bf tv = *(const v4bf*)(t + e*DD + lane*4);
  v4f  pv = *(const v4f*)(P + (long)d*DD + lane*4);
  v4f  tf = __builtin_convertvector(tv, v4f);
  float s = tf[0]*pv[0] + tf[1]*pv[1] + tf[2]*pv[2] + tf[3]*pv[3];
#pragma unroll
  for (int o = 16; o > 0; o >>= 1) s += __shfl_xor(s, o, 32);
  if (lane == 0){
    float a = (s + c[d]) * etype[e] * 0.08838834764831845f;   // 1/sqrt(128)
    attn[e] = a;
    atomicMax(nmax + d, fenc(a));
  }
}

// ------- kernel 6: e = exp(attn - m[dst]); segment sum -------
__global__ __launch_bounds__(256)
void edge_exp_kernel(float* __restrict__ attn, const int* __restrict__ dst,
                     const unsigned* __restrict__ nmax, float* __restrict__ nsum){
  long e = (long)blockIdx.x * 256 + threadIdx.x;
  if (e >= N_EDGES) return;
  int d = dst[e];
  float w = expf(attn[e] - fdec(nmax[d]));
  attn[e] = w;
  atomicAdd(nsum + d, w);
}

// ------- kernel 7: u[dst] += a_e * t_e -------
__global__ __launch_bounds__(256)
void edge_scatter_kernel(const __bf16* __restrict__ t, const float* __restrict__ attn,
                         const int* __restrict__ dst, const float* __restrict__ nsum,
                         float* __restrict__ u){
  int wave = threadIdx.x >> 5, lane = threadIdx.x & 31;
  long e = (long)blockIdx.x * 8 + wave;
  if (e >= N_EDGES) return;
  int d = dst[e];
  float a = attn[e] / (nsum[d] + 1e-16f);
  v4bf tv = *(const v4bf*)(t + e*DD + lane*4);
  v4f  tf = __builtin_convertvector(tv, v4f);
  float* ur = u + (long)d*DD + lane*4;
#pragma unroll
  for (int i = 0; i < 4; ++i) atomicAdd(ur + i, a * tf[i]);
}

// ------- kernel 8: out = x + gelu(u @ Wv + sa*bv), sa = s/(s+eps) -------
__global__ __launch_bounds__(256)
void node_out_kernel(const float* __restrict__ u, const __bf16* __restrict__ wvT,
                     const float* __restrict__ bv, const float* __restrict__ nsum,
                     const float* __restrict__ x, float* __restrict__ out){
  extern __shared__ __align__(16) char smem[];
  __bf16* Wv_l = (__bf16*)smem;                           // 128 rows * W1_STR
  float*  Sbase = (float*)(smem + 128*W1_STR*2);

  int wave = threadIdx.x >> 5, lane = threadIdx.x & 31;
  int tid = threadIdx.x;
  if (tid < 128) copy_row_to_lds(Wv_l + tid*W1_STR, wvT + tid*DD, DD);
  __syncthreads();

  int m = lane & 15, half = lane >> 4;
  float* L = Sbase + wave * 16*SF_STR;

  for (int tile = blockIdx.x*8 + wave; tile < N_NODES/16; tile += gridDim.x*8){
    long nb = (long)tile * 16;
    const float* urow = u + (nb + m) * DD;

    v8f acc[8] = {};
#pragma unroll
    for (int kt = 0; kt < 4; ++kt){
      int k0 = kt * 32;
      v16bf a = fragA_f32(urow, k0, half);
#pragma unroll
      for (int nt = 0; nt < 8; ++nt)
        acc[nt] = wmma_bf16(a, fragB_bf(Wv_l + (nt*16 + m)*W1_STR, k0, half), acc[nt]);
    }
    float sa[8];
#pragma unroll
    for (int r = 0; r < 8; ++r){
      float s = nsum[nb + r + 8*half];
      sa[r] = s / (s + 1e-16f);                   // = segment_sum of a_e
    }
#pragma unroll
    for (int nt = 0; nt < 8; ++nt){
      int col = nt*16 + m; float bvc = bv[col];
#pragma unroll
      for (int r = 0; r < 8; ++r)
        L[(r + 8*half)*SF_STR + col] = gelu_exact(acc[nt][r] + sa[r]*bvc);
    }
    __builtin_amdgcn_wave_barrier();
    const float* xrow = x + nb * DD;
    float* orow = out + nb * DD;
#pragma unroll
    for (int i = 0; i < 16; ++i){
      v4f g  = *(const v4f*)(L + i*SF_STR + lane*4);
      v4f xv = *(const v4f*)(xrow + i*DD + lane*4);
      g += xv;
      *(v4f*)(orow + i*DD + lane*4) = g;
    }
    __builtin_amdgcn_wave_barrier();
  }
}

extern "C" void kernel_launch(void* const* d_in, const int* in_sizes, int n_in,
                              void* d_out, int out_size, void* d_ws, size_t ws_size,
                              hipStream_t stream) {
  const float* x     = (const float*)d_in[0];
  const int*   ei    = (const int*)  d_in[1];   // [2, E]
  const float* etype = (const float*)d_in[2];
  const float* ev    = (const float*)d_in[3];
  const float* gamma = (const float*)d_in[4];
  const float* beta  = (const float*)d_in[5];
  const float* Wt    = (const float*)d_in[6];
  const float* bt    = (const float*)d_in[7];
  const float* Wq    = (const float*)d_in[8];
  const float* bq    = (const float*)d_in[9];
  const float* Wk    = (const float*)d_in[10];
  const float* bk    = (const float*)d_in[11];
  const float* Wv    = (const float*)d_in[12];
  const float* bv    = (const float*)d_in[13];
  const int* src  = ei;
  const int* dstI = ei + N_EDGES;
  float* out = (float*)d_out;

  char* ws = (char*)d_ws;
  size_t off = 0;
  auto alloc = [&](size_t bytes) -> void* {
    void* p = ws + off; off += (bytes + 255) & ~(size_t)255; return p;
  };
  __bf16* hb   = (__bf16*)alloc((size_t)N_NODES * DD * 2);
  __bf16* wtT  = (__bf16*)alloc((size_t)128 * 256 * 2);
  __bf16* wqT  = (__bf16*)alloc((size_t)128 * 128 * 2);
  __bf16* wkb  = (__bf16*)alloc((size_t)128 * 128 * 2);
  __bf16* wvT  = (__bf16*)alloc((size_t)128 * 128 * 2);
  float*  P    = (float*) alloc((size_t)N_NODES * DD * 4);
  float*  c    = (float*) alloc((size_t)N_NODES * 4);
  __bf16* t    = (__bf16*)alloc((size_t)N_EDGES * DD * 2);
  float*  attn = (float*) alloc((size_t)N_EDGES * 4);
  char* zbase = ws + off;                        // contiguous zero region
  float*    u    = (float*)   alloc((size_t)N_NODES * DD * 4);
  float*    nsum = (float*)   alloc((size_t)N_NODES * 4);
  unsigned* nmax = (unsigned*)alloc((size_t)N_NODES * 4);
  size_t zbytes = (size_t)((ws + off) - zbase);

  hipMemsetAsync(zbase, 0, zbytes, stream);      // u=0, nsum=0, nmax=enc-min

  prep_w_kernel<<<(128*256 + 128*128 + 255) / 256, 256, 0, stream>>>(
      Wt, Wq, Wk, Wv, wtT, wqT, wkb, wvT);
  ln_kernel<<<(N_NODES + 7) / 8, 256, 0, stream>>>(x, gamma, beta, hb);

  size_t qp_lds  = (size_t)2*128*W1_STR*2 + (size_t)8*16*SF_STR*4;   // ~137 KB
  node_qp_kernel<<<256, 256, qp_lds, stream>>>(hb, wqT, bq, wkb, bk, P, c);

  size_t et_lds  = (size_t)128*WT_STR*2 + (size_t)8*16*SB_STR*2;     // ~100 KB
  edge_t_kernel<<<1024, 256, et_lds, stream>>>(hb, ev, src, wtT, bt, t);

  edge_attn_kernel<<<(N_EDGES + 7) / 8, 256, 0, stream>>>(t, P, c, dstI, etype, attn, nmax);
  edge_exp_kernel<<<(N_EDGES + 255) / 256, 256, 0, stream>>>(attn, dstI, nmax, nsum);
  edge_scatter_kernel<<<(N_EDGES + 7) / 8, 256, 0, stream>>>(t, attn, dstI, nsum, u);

  size_t no_lds  = (size_t)128*W1_STR*2 + (size_t)8*16*SF_STR*4;     // ~102 KB
  node_out_kernel<<<256, 256, no_lds, stream>>>(u, wvT, bv, nsum, x, out);

  (void)in_sizes; (void)n_in; (void)out_size; (void)ws_size;
}
